// GlmMoeDsaIndexer_22960895164469
// MI455X (gfx1250) — matrix operand order repaired and checked by reading, not compiled, not run on hardware
//
#include <hip/hip_runtime.h>

// ---------------------------------------------------------------------------
// GLM MoE DSA indexer for gfx1250 (CDNA5, wave32, WMMA bf16).
// Pipeline: f32->bf16 convert | (k,weights) WMMA GEMM | LN+RoPE(k) |
//           fused per-head q WMMA GEMM (A staged to LDS via async-to-LDS)
//           + RoPE + weighted head-sum | scores WMMA GEMM + mask |
//           per-row bitonic top-512.
// ---------------------------------------------------------------------------

typedef __bf16 bf16;
typedef __attribute__((ext_vector_type(8)))  __bf16 bfx8;
typedef __attribute__((ext_vector_type(16))) __bf16 bfx16;
typedef __attribute__((ext_vector_type(8)))  float  v8f;

#define S_LEN 2048
#define HID   4096
#define NH    32
#define DH    128
#define RR    64
#define QLEN  1536
#define TOPK  512

static __device__ __forceinline__ v8f zero8() {
  v8f r;
#pragma unroll
  for (int i = 0; i < 8; ++i) r[i] = 0.0f;
  return r;
}

static __device__ __forceinline__ bfx16 cat8(bfx8 lo, bfx8 hi) {
  bfx16 r;
#pragma unroll
  for (int i = 0; i < 8; ++i) { r[i] = lo[i]; r[i + 8] = hi[i]; }
  return r;
}

// A-matrix 16x32 bf16 fragment: lanes 0-15 hold row M=lane,
// K = base+0..15 (contiguous, matching B); lanes 16-31 hold the same
// rows with K = base+16..31.  rowkp = &A[row][kbase].
static __device__ __forceinline__ bfx16 load_afrag(const bf16* rowkp, int lhi) {
  const bf16* p = rowkp + lhi * 16;
  bfx8 lo = *(const bfx8*)(p);
  bfx8 hi = *(const bfx8*)(p + 8);
  return cat8(lo, hi);
}

static __device__ __forceinline__ v8f wmma_bf16(bfx16 a, bfx16 b, v8f c) {
  return __builtin_amdgcn_wmma_f32_16x16x32_bf16(false, a, false, b,
                                                 (short)0, c, false, false);
}

// ---------------------------------------------------------------------------
// Kernel 1: f32 -> bf16 conversion, 4 elements / thread (b128 loads).
// ---------------------------------------------------------------------------
__global__ void f2bf_kernel(const float* __restrict__ in, bf16* __restrict__ out,
                            int n) {
  int i = (blockIdx.x * blockDim.x + threadIdx.x) * 4;
  if (i + 3 < n) {
    float4 v = *(const float4*)(in + i);
    out[i + 0] = (bf16)v.x;
    out[i + 1] = (bf16)v.y;
    out[i + 2] = (bf16)v.z;
    out[i + 3] = (bf16)v.w;
  }
}

// ---------------------------------------------------------------------------
// Kernel 2: k_raw = hid(2048x4096) @ wk_w^T (->128 cols) and
//           wsc   = hid @ wproj^T * (1/64)   (->32 cols, scale folded)
// 2 waves per 16-row stripe; each wave owns 5 of the 10 N-tiles.
// B fragments are software-pipelined one tile ahead; A one K-step ahead.
// ---------------------------------------------------------------------------
__global__ __launch_bounds__(64)
void kw_gemm_kernel(const bf16* __restrict__ hid, const bf16* __restrict__ wkw,
                    const bf16* __restrict__ wpj, float* __restrict__ kraw,
                    float* __restrict__ wsc) {
  const int m0   = blockIdx.x * 16;
  const int wave = threadIdx.x >> 5;
  const int lane = threadIdx.x & 31;
  const int lrow = lane & 15, lhi = lane >> 4;
  const int t0   = wave * 5;  // this wave's first N-tile (5 tiles each)

  auto loadB = [&](int t, int k) -> bfx16 {
    // B-matrix 32x16: lane holds column N=lane%16, 16 contiguous K values
    // starting at k + 16*(lane/16).  B[k][n] = W[n][k] (W row-major NxK).
    const bf16* brow = (t < 8) ? (wkw + (size_t)(t * 16 + lrow) * HID)
                               : (wpj + (size_t)((t - 8) * 16 + lrow) * HID);
    return *(const bfx16*)(brow + k + lhi * 16);
  };

  v8f acc[5];
#pragma unroll
  for (int tt = 0; tt < 5; ++tt) acc[tt] = zero8();

  const bf16* arow = hid + (size_t)(m0 + lrow) * HID;
  bfx16 a_cur = load_afrag(arow, lhi);
  bfx16 bcur  = loadB(t0, 0);
  for (int kk = 0; kk < HID; kk += 32) {
    const int kn = (kk + 32 < HID) ? kk + 32 : kk;
    bfx16 a_next = load_afrag(arow + kn, lhi);
#pragma unroll
    for (int tt = 0; tt < 5; ++tt) {
      bfx16 bnext = (tt < 4) ? loadB(t0 + tt + 1, kk) : loadB(t0, kn);
      acc[tt] = wmma_bf16(a_cur, bcur, acc[tt]);
      bcur = bnext;
    }
    a_cur = a_next;
  }
  // C/D layout: N = lane%16, VGPR i -> M = i + 8*(lane/16)
#pragma unroll
  for (int tt = 0; tt < 5; ++tt) {
    const int t = t0 + tt;
    if (t < 8) {
      int col = t * 16 + lrow;
#pragma unroll
      for (int i = 0; i < 8; ++i)
        kraw[(size_t)(m0 + i + 8 * lhi) * DH + col] = acc[tt][i];
    } else {
      int col = (t - 8) * 16 + lrow;
#pragma unroll
      for (int i = 0; i < 8; ++i)
        wsc[(size_t)(m0 + i + 8 * lhi) * NH + col] = acc[tt][i] * 0.015625f;
    }
  }
}

// ---------------------------------------------------------------------------
// Kernel 3: layernorm over D=128 + RoPE on first 64 dims of k, emit bf16.
// ---------------------------------------------------------------------------
__global__ __launch_bounds__(128)
void k_ln_rope_kernel(const float* __restrict__ kraw, const float* __restrict__ cosp,
                      const float* __restrict__ sinp, const float* __restrict__ gw,
                      const float* __restrict__ gb, bf16* __restrict__ kb) {
  const int s = blockIdx.x, d = threadIdx.x;
  __shared__ float red[DH];
  __shared__ float row[DH];
  float x = kraw[(size_t)s * DH + d];
  red[d] = x;
  __syncthreads();
#pragma unroll
  for (int off = 64; off > 0; off >>= 1) {
    if (d < off) red[d] += red[d + off];
    __syncthreads();
  }
  float mu = red[0] * (1.0f / DH);
  __syncthreads();
  float dx = x - mu;
  red[d] = dx * dx;
  __syncthreads();
#pragma unroll
  for (int off = 64; off > 0; off >>= 1) {
    if (d < off) red[d] += red[d + off];
    __syncthreads();
  }
  float var = red[0] * (1.0f / DH);
  float y = dx * rsqrtf(var + 1e-6f) * gw[d] + gb[d];
  row[d] = y;
  __syncthreads();
  float out = y;
  if (d < RR) {
    float rot = (d < RR / 2) ? -row[d + RR / 2] : row[d - RR / 2];
    out = y * cosp[(size_t)s * RR + d] + rot * sinp[(size_t)s * RR + d];
  }
  kb[(size_t)s * DH + d] = (bf16)out;
}

// ---------------------------------------------------------------------------
// Kernel 4: fused q = q_resid @ wq_b^T, per-head RoPE, scale by per-(s,h)
// weight (1/64 folded), sum over heads -> qsum bf16 (2048x128).
// The 16x1536 A-stripe (48 KB, contiguous rows) is staged to LDS ONCE per
// workgroup with global_load_async_to_lds_b128 (ASYNCcnt), then every wave /
// head reads A fragments via ds_load.  B fragments pipelined one tile ahead.
// Static LDS = 48K (A) + 4K (cos) + 4K (sin) + 8K (head-sum) = 64 KB.
// ---------------------------------------------------------------------------
__global__ __launch_bounds__(128)
void q_fused_kernel(const bf16* __restrict__ qres, const bf16* __restrict__ wqb,
                    const float* __restrict__ wsc, const float* __restrict__ cosp,
                    const float* __restrict__ sinp, bf16* __restrict__ qsum) {
  const int m0   = blockIdx.x * 16;
  const int tid  = threadIdx.x;
  const int wave = tid >> 5, lane = tid & 31;
  const int lrow = lane & 15, lhi = lane >> 4;

  __shared__ bf16  s_a[16 * QLEN];   // 48 KB A stripe
  __shared__ float s_cos[16 * RR];   // 4 KB
  __shared__ float s_sin[16 * RR];   // 4 KB
  __shared__ float s_q[16 * DH];     // 8 KB head-sum accumulator

  // --- async stage of the contiguous 48 KB A stripe into LDS -------------
  {
    const char* gsrc  = (const char*)(qres + (size_t)m0 * QLEN);
    unsigned    lbase = (unsigned)(size_t)(&s_a[0]);
    for (int c = tid; c < (16 * QLEN * 2) / 16; c += 128) {
      unsigned    loff = lbase + (unsigned)c * 16u;
      const char* g    = gsrc + (size_t)c * 16;
      asm volatile("global_load_async_to_lds_b128 %0, %1, off"
                   :: "v"(loff), "v"(g)
                   : "memory");
    }
  }
  // fill cos/sin/head-sum while the async DMA is in flight
  for (int i = tid; i < 16 * RR; i += 128) {
    int m = i >> 6, c = i & 63;
    s_cos[i] = cosp[(size_t)(m0 + m) * RR + c];
    s_sin[i] = sinp[(size_t)(m0 + m) * RR + c];
  }
  for (int i = tid; i < 16 * DH; i += 128) s_q[i] = 0.0f;
  asm volatile("s_wait_asynccnt 0x0" ::: "memory");
  __syncthreads();

  float qa[8][8];
#pragma unroll
  for (int t = 0; t < 8; ++t)
#pragma unroll
    for (int i = 0; i < 8; ++i) qa[t][i] = 0.0f;

  auto loadB = [&](int h, int t, int k) -> bfx16 {
    return *(const bfx16*)(wqb + (size_t)(h * DH + t * 16 + lrow) * QLEN +
                           k + lhi * 16);
  };

  const bf16* arow = s_a + (size_t)lrow * QLEN;  // LDS, ds_load_b128 fragments
  for (int hh = 0; hh < 8; ++hh) {
    const int h = wave + hh * 4;  // heads interleaved across the 4 waves
    v8f acc[8];
#pragma unroll
    for (int t = 0; t < 8; ++t) acc[t] = zero8();

    bfx16 bcur = loadB(h, 0, 0);
    for (int kk = 0; kk < QLEN; kk += 32) {
      const int kn = (kk + 32 < QLEN) ? kk + 32 : kk;
      bfx16 a = load_afrag(arow + kk, lhi);
#pragma unroll
      for (int t = 0; t < 8; ++t) {
        bfx16 bnext = (t < 7) ? loadB(h, t + 1, kk) : loadB(h, 0, kn);
        acc[t] = wmma_bf16(a, bcur, acc[t]);
        bcur = bnext;
      }
    }
    // Epilogue: RoPE on cols 0..63.  Col c's rotation partner c^32 sits in
    // acc tile t^2 at the same VGPR index (same lane, same i).
#pragma unroll
    for (int i = 0; i < 8; ++i) {
      const int m = i + 8 * lhi;                              // local row
      const float wl = wsc[(size_t)(m0 + m) * NH + h];        // w*(H*D)^-0.5
#pragma unroll
      for (int t = 0; t < 4; ++t) {
        const int c = t * 16 + lrow;
        float rot = (t < 2) ? -acc[t + 2][i] : acc[t - 2][i];
        float nv = acc[t][i] * s_cos[m * RR + c] + rot * s_sin[m * RR + c];
        qa[t][i] += nv * wl;
      }
#pragma unroll
      for (int t = 4; t < 8; ++t) qa[t][i] += acc[t][i] * wl;
    }
  }
  // Cross-wave head-sum via LDS float atomics (ds_add_f32).
#pragma unroll
  for (int t = 0; t < 8; ++t) {
    const int c = t * 16 + lrow;
#pragma unroll
    for (int i = 0; i < 8; ++i)
      atomicAdd(&s_q[(i + 8 * lhi) * DH + c], qa[t][i]);
  }
  __syncthreads();
  for (int i = tid; i < 16 * DH; i += 128)
    qsum[(size_t)(m0 + (i >> 7)) * DH + (i & 127)] = (bf16)s_q[i];
}

// ---------------------------------------------------------------------------
// Kernel 5: scores = qsum(2048x128) @ k^T + mask.  K=128 -> 4 WMMA steps,
// A fragments hoisted; B fragments pipelined one step ahead.
// ---------------------------------------------------------------------------
__global__ __launch_bounds__(128)
void score_gemm_kernel(const bf16* __restrict__ qsb, const bf16* __restrict__ kb,
                       const float* __restrict__ mask, float* __restrict__ scores) {
  const int m0   = blockIdx.x * 16;
  const int tid  = threadIdx.x;
  const int wave = tid >> 5, lane = tid & 31;
  const int lrow = lane & 15, lhi = lane >> 4;

  bfx16 afr[4];
  const bf16* arow = qsb + (size_t)(m0 + lrow) * DH;
#pragma unroll
  for (int ks = 0; ks < 4; ++ks) afr[ks] = load_afrag(arow + ks * 32, lhi);

  auto loadB = [&](int tn, int ks) -> bfx16 {
    return *(const bfx16*)(kb + (size_t)(tn * 16 + lrow) * DH +
                           ks * 32 + lhi * 16);
  };

  const int tn0 = wave * 32;
  bfx16 bcur = loadB(tn0, 0);
  for (int tn = tn0; tn < tn0 + 32; ++tn) {
    v8f acc = zero8();
#pragma unroll
    for (int ks = 0; ks < 4; ++ks) {
      const int nt = (ks < 3) ? tn : ((tn + 1 < tn0 + 32) ? tn + 1 : tn);
      const int nk = (ks < 3) ? ks + 1 : 0;
      bfx16 bnext = loadB(nt, nk);
      acc = wmma_bf16(afr[ks], bcur, acc);
      bcur = bnext;
    }
    const int n = tn * 16 + lrow;
#pragma unroll
    for (int i = 0; i < 8; ++i) {
      const size_t srow = (size_t)(m0 + i + 8 * lhi);
      scores[srow * S_LEN + n] = acc[i] + mask[srow * S_LEN + n];
    }
  }
}

// ---------------------------------------------------------------------------
// Kernel 6: per-row top-512.  Pack (sortable_f32<<32)|(~index) into 64-bit
// keys, bitonic-sort 2048 keys descending in LDS (matches lax.top_k order:
// value desc, index asc on ties), emit int32 indices.
// ---------------------------------------------------------------------------
__global__ __launch_bounds__(256)
void topk_kernel(const float* __restrict__ scores, int* __restrict__ out) {
  const int s = blockIdx.x, tid = threadIdx.x;
  __shared__ unsigned long long keys[S_LEN];  // 16 KB

  for (int j = tid; j < S_LEN; j += 256) {
    unsigned int b = __float_as_uint(scores[(size_t)s * S_LEN + j]);
    unsigned int u = b ^ ((unsigned int)(((int)b) >> 31) | 0x80000000u);
    keys[j] = ((unsigned long long)u << 32) |
              (unsigned long long)(0xFFFFFFFFu - (unsigned int)j);
  }
  __syncthreads();

  for (unsigned int k = 2; k <= (unsigned int)S_LEN; k <<= 1) {
    for (unsigned int jj = k >> 1; jj > 0; jj >>= 1) {
      for (unsigned int i = tid; i < (unsigned int)S_LEN; i += 256) {
        unsigned int ixj = i ^ jj;
        if (ixj > i) {
          unsigned long long a = keys[i], b = keys[ixj];
          bool doswap = ((i & k) == 0) ? (a < b) : (a > b);  // descending
          if (doswap) { keys[i] = b; keys[ixj] = a; }
        }
      }
      __syncthreads();
    }
  }
  for (int t = tid; t < TOPK; t += 256) {
    unsigned int low = (unsigned int)(keys[t] & 0xFFFFFFFFull);
    out[(size_t)s * TOPK + t] = (int)(0xFFFFFFFFu - low);
  }
}

// ---------------------------------------------------------------------------
// Launcher
// ---------------------------------------------------------------------------
extern "C" void kernel_launch(void* const* d_in, const int* in_sizes, int n_in,
                              void* d_out, int out_size, void* d_ws, size_t ws_size,
                              hipStream_t stream) {
  (void)in_sizes; (void)n_in; (void)out_size; (void)ws_size;
  const float* hidden = (const float*)d_in[0];   // (1,2048,4096)
  const float* qresid = (const float*)d_in[1];   // (1,2048,1536)
  const float* cosp   = (const float*)d_in[2];   // (1,2048,64)
  const float* sinp   = (const float*)d_in[3];   // (1,2048,64)
  const float* mask   = (const float*)d_in[4];   // (1,2048,2048)
  const float* wq_b   = (const float*)d_in[5];   // (4096,1536)
  const float* wk_w   = (const float*)d_in[6];   // (128,4096)
  const float* knw    = (const float*)d_in[7];   // (128,)
  const float* knb    = (const float*)d_in[8];   // (128,)
  const float* wproj  = (const float*)d_in[9];   // (32,4096)
  int* out = (int*)d_out;

  char* ws = (char*)d_ws;
  size_t o = 0;
  bf16*  hidb = (bf16*)(ws + o);  o += (size_t)S_LEN * HID * 2;          // 16 MB
  bf16*  qrb  = (bf16*)(ws + o);  o += (size_t)S_LEN * QLEN * 2;         //  6 MB
  bf16*  wqbb = (bf16*)(ws + o);  o += (size_t)NH * DH * QLEN * 2;       // 12 MB
  bf16*  wkwb = (bf16*)(ws + o);  o += (size_t)DH * HID * 2;             //  1 MB
  bf16*  wpjb = (bf16*)(ws + o);  o += (size_t)NH * HID * 2;             // .25MB
  float* kraw = (float*)(ws + o); o += (size_t)S_LEN * DH * 4;           //  1 MB
  float* wsc  = (float*)(ws + o); o += (size_t)S_LEN * NH * 4;           // .25MB
  bf16*  kb   = (bf16*)(ws + o);  o += (size_t)S_LEN * DH * 2;           // .5 MB
  bf16*  qsb  = (bf16*)(ws + o);  o += (size_t)S_LEN * DH * 2;           // .5 MB
  float* scr  = (float*)(ws + o); o += (size_t)S_LEN * S_LEN * 4;        // 16 MB

  auto cvt = [&](const float* src, bf16* dst, int n) {
    f2bf_kernel<<<(n / 4 + 255) / 256, 256, 0, stream>>>(src, dst, n);
  };
  cvt(hidden, hidb, S_LEN * HID);
  cvt(qresid, qrb, S_LEN * QLEN);
  cvt(wq_b, wqbb, NH * DH * QLEN);
  cvt(wk_w, wkwb, DH * HID);
  cvt(wproj, wpjb, NH * HID);

  kw_gemm_kernel<<<S_LEN / 16, 64, 0, stream>>>(hidb, wkwb, wpjb, kraw, wsc);
  k_ln_rope_kernel<<<S_LEN, 128, 0, stream>>>(kraw, cosp, sinp, knw, knb, kb);
  q_fused_kernel<<<S_LEN / 16, 128, 0, stream>>>(qrb, wqbb, wsc, cosp, sinp, qsb);
  score_gemm_kernel<<<S_LEN / 16, 128, 0, stream>>>(qsb, kb, mask, scr);
  topk_kernel<<<S_LEN, 256, 0, stream>>>(scr, out);
}
